// FeedForward_83803401879881
// MI455X (gfx1250) — 8-bit matrix operand order repaired and checked by reading, compiled, not run on hardware
//
#include <hip/hip_runtime.h>
#include <stddef.h>


typedef int v8i __attribute__((ext_vector_type(8)));

#define D_MODEL 512
#define D_FF    2048

#define TOK1    32
#define SXS     520
#define SHS     2056

#define TOK2    128
#define KC      512
#define CSTR    520

__device__ __forceinline__ void async_ld16(unsigned lds_off, const void* gptr) {
    asm volatile("global_load_async_to_lds_b128 %0, %1, off"
                 :: "v"(lds_off), "v"((unsigned long long)(size_t)gptr)
                 : "memory");
}
#define WAIT_ASYNC(n) asm volatile("s_wait_asynccnt " #n ::: "memory")

__global__ __launch_bounds__(256) void wquant_kernel(
    const float* __restrict__ w_up, const float* __restrict__ w_down,
    signed char* __restrict__ wq_up, signed char* __restrict__ wq_down,
    float* __restrict__ wscales)
{
    const int n = D_MODEL * D_FF;
    const float*  w  = (blockIdx.x == 0) ? w_up  : w_down;
    signed char*  wq = (blockIdx.x == 0) ? wq_up : wq_down;

    __shared__ float sred[256];
    __shared__ float s_scale;
    const int tid = threadIdx.x;

    float s = 0.0f;
    for (int i = tid; i < n; i += 256) s += fabsf(w[i]);
    sred[tid] = s;
    __syncthreads();
    for (int off = 128; off > 0; off >>= 1) {
        if (tid < off) sred[tid] += sred[tid + off];
        __syncthreads();
    }
    if (tid == 0) {
        float sc = fmaxf(sred[0] / (float)n, 1e-5f);
        s_scale = sc;
        wscales[blockIdx.x] = sc;
    }
    __syncthreads();
    const float inv = 1.0f / s_scale;
    for (int i = tid; i < n; i += 256) {
        float q = rintf(w[i] * inv);
        q = fminf(fmaxf(q, -1.0f), 1.0f);
        wq[i] = (signed char)(int)q;
    }
}

__global__ __launch_bounds__(256) void gemm1_kernel(
    const float* __restrict__ x,
    const signed char* __restrict__ wq_up,
    const float* __restrict__ wscales,
    signed char* __restrict__ h2q,
    float* __restrict__ rowmax_out)
{
    __shared__ signed char sx[TOK1 * SXS];
    __shared__ signed char sh2[TOK1 * SHS];
    __shared__ int   s_absmax[TOK1];
    __shared__ float s_xs[TOK1];
    __shared__ float s_hscale[TOK1];
    __shared__ int   s_rowmax[TOK1];
    __shared__ float s_rq[TOK1];

    const int tid  = threadIdx.x;
    const int tok0 = blockIdx.x * TOK1;

    if (tid < TOK1) { s_absmax[tid] = 0; s_rowmax[tid] = 0; }
    __syncthreads();

    for (int i = tid; i < TOK1 * D_MODEL; i += 256) {
        float v = fabsf(x[(size_t)tok0 * D_MODEL + i]);
        atomicMax(&s_absmax[i >> 9], __float_as_int(v));
    }
    __syncthreads();
    if (tid < TOK1) {
        float mx = fmaxf(__int_as_float(s_absmax[tid]), 1e-5f);
        s_xs[tid]     = 127.0f / mx;
        s_hscale[tid] = wscales[0] * mx * (1.0f / 127.0f);
    }
    __syncthreads();

    for (int i = tid; i < TOK1 * D_MODEL; i += 256) {
        int trow = i >> 9, k = i & 511;
        float q = rintf(x[(size_t)tok0 * D_MODEL + i] * s_xs[trow]);
        q = fminf(fmaxf(q, -128.0f), 127.0f);
        sx[trow * SXS + k] = (signed char)(int)q;
    }
    __syncthreads();

    const int wave = tid >> 5, lane = tid & 31;
    const int lrow = lane & 15;
    const int half = lane >> 4;

    for (int nb = 0; nb < D_FF; nb += 8 * 16) {
        const int col = nb + wave * 16 + lrow;

        v8i B[8];
#pragma unroll
        for (int kb = 0; kb < 8; ++kb)
#pragma unroll
            for (int v = 0; v < 8; ++v) {
                int k = kb * 64 + (v >> 2) * 32 + half * 16 + (v & 3) * 4;
                B[kb][v] = *(const int*)(wq_up + (size_t)col * D_MODEL + k);
            }

#pragma unroll
        for (int m = 0; m < TOK1 / 16; ++m) {
            v8i acc = {};
            const int arow = m * 16 + lrow;
#pragma unroll
            for (int kb = 0; kb < 8; ++kb) {
                v8i A;
#pragma unroll
                for (int v = 0; v < 8; ++v) {
                    int k = kb * 64 + (v >> 2) * 32 + half * 16 + (v & 3) * 4;
                    A[v] = *(const int*)(sx + arow * SXS + k);
                }
                acc = __builtin_amdgcn_wmma_i32_16x16x64_iu8(
                          true, A, true, B[kb], acc, false, false);
            }
            const int rowbase = m * 16 + half * 8;
#pragma unroll
            for (int r = 0; r < 8; ++r) {
                const int trow = rowbase + r;
                float h = fmaxf((float)acc[r] * s_hscale[trow], 0.0f);
                int qi = (int)fminf(rintf(h), 127.0f);
                sh2[trow * SHS + col] = (signed char)qi;
            }
        }
    }
    __syncthreads();

    {
        const int trow = tid >> 3, k0 = (tid & 7) * 256;
        int m = 0;
#pragma unroll 8
        for (int j = 0; j < 64; ++j) {
            int raw = *(const int*)(sh2 + trow * SHS + k0 + j * 4);
            m = max(m, raw & 255); m = max(m, (raw >> 8) & 255);
            m = max(m, (raw >> 16) & 255); m = max(m, (raw >> 24) & 255);
        }
        atomicMax(&s_rowmax[trow], m);
    }
    __syncthreads();
    if (tid < TOK1) {
        float mx = (float)s_rowmax[tid];
        rowmax_out[tok0 + tid] = mx;
        s_rq[tid] = 127.0f / fmaxf(mx, 1e-5f);
    }
    __syncthreads();

    for (int it = 0; it < (TOK1 * D_FF / 4) / 256; ++it) {
        const int i4   = (tid + it * 256) * 4;
        const int trow = i4 >> 11, k = i4 & (D_FF - 1);
        int raw = *(const int*)(sh2 + trow * SHS + k);
        const float sc = s_rq[trow];
        int packed = 0;
#pragma unroll
        for (int b = 0; b < 4; ++b) {
            int v = (raw >> (8 * b)) & 255;
            int q = (int)fminf(rintf((float)v * sc), 127.0f);
            packed |= (q & 255) << (8 * b);
        }
        *(int*)(h2q + (size_t)(tok0 + trow) * D_FF + k) = packed;
    }
}

__device__ __forceinline__ void stage_chunk(unsigned lds_base,
                                            const signed char* __restrict__ h2q,
                                            int tok0, int kc, int tid)
{
#pragma unroll
    for (int it = 0; it < 16; ++it) {
        const int i    = tid + it * 256;
        const int trow = i >> 5;
        const int koff = (i & 31) * 16;
        async_ld16(lds_base + (unsigned)(trow * CSTR + koff),
                   h2q + (size_t)(tok0 + trow) * D_FF + kc + koff);
    }
}

__global__ __launch_bounds__(256) void gemm2_kernel(
    const signed char* __restrict__ h2q,
    const signed char* __restrict__ wq_down,
    const float* __restrict__ wscales,
    const float* __restrict__ rowmax,
    float* __restrict__ out)
{
    __shared__ signed char sh[2][TOK2 * CSTR];
    __shared__ float s_os[TOK2];

    const int tid  = threadIdx.x;
    const int tok0 = blockIdx.x * TOK2;

    if (tid < TOK2)
        s_os[tid] = wscales[1] * fmaxf(rowmax[tok0 + tid], 1e-5f) * (1.0f / 127.0f);
    __syncthreads();

    const unsigned sb0 = (unsigned)(size_t)&sh[0][0];
    const unsigned sb1 = (unsigned)(size_t)&sh[1][0];

    const int wave = tid >> 5, lane = tid & 31;
    const int lrow = lane & 15;
    const int half = lane >> 4;

    for (int nb = 0; nb < D_MODEL; nb += 8 * 16) {
        const int col = nb + wave * 16 + lrow;
        v8i acc[TOK2 / 16] = {};

        stage_chunk(sb0, h2q, tok0, 0, tid);

        for (int c = 0; c < D_FF / KC; ++c) {
            const int kc = c * KC;
            const unsigned sbc = (c & 1) ? sb1 : sb0;
            signed char* shc = (c & 1) ? sh[1] : sh[0];

            if (c < 3) {
                stage_chunk((c & 1) ? sb0 : sb1, h2q, tok0, kc + KC, tid);
                WAIT_ASYNC(0x10);
            } else {
                WAIT_ASYNC(0x0);
            }
            __syncthreads();

#pragma unroll
            for (int kk = 0; kk < KC / 64; ++kk) {
                v8i B;
#pragma unroll
                for (int v = 0; v < 8; ++v) {
                    int k = kc + kk * 64 + (v >> 2) * 32 + half * 16 + (v & 3) * 4;
                    B[v] = *(const int*)(wq_down + (size_t)col * D_FF + k);
                }
#pragma unroll
                for (int m = 0; m < TOK2 / 16; ++m) {
                    v8i A;
#pragma unroll
                    for (int v = 0; v < 8; ++v) {
                        int k = kk * 64 + (v >> 2) * 32 + half * 16 + (v & 3) * 4;
                        A[v] = *(const int*)(shc + (m * 16 + lrow) * CSTR + k);
                    }
                    acc[m] = __builtin_amdgcn_wmma_i32_16x16x64_iu8(
                                 true, A, true, B, acc[m], false, false);
                }
            }
            __syncthreads();
        }

#pragma unroll
        for (int m = 0; m < TOK2 / 16; ++m) {
            const int rowbase = m * 16 + half * 8;
#pragma unroll
            for (int r = 0; r < 8; ++r) {
                const int trow = rowbase + r;
                out[(size_t)(tok0 + trow) * D_MODEL + col] =
                    (float)acc[m][r] * s_os[trow];
            }
        }
    }
}

extern "C" void kernel_launch(void* const* d_in, const int* in_sizes, int n_in,
                              void* d_out, int out_size, void* d_ws, size_t ws_size,
                              hipStream_t stream) {
    const float* x      = (const float*)d_in[0];
    const float* w_up   = (const float*)d_in[1];
    const float* w_down = (const float*)d_in[2];
    float* out = (float*)d_out;

    char* ws = (char*)d_ws;
    signed char* wq_up   = (signed char*)(ws);
    signed char* wq_down = (signed char*)(ws + (1 << 20));
    float*       wscales = (float*)      (ws + (2 << 20));
    float*       rowmax  = (float*)      (ws + (2 << 20) + 1024);
    signed char* h2q     = (signed char*)(ws + (4 << 20));

    const int tokens = in_sizes[0] / D_MODEL;

    wquant_kernel<<<2, 256, 0, stream>>>(w_up, w_down, wq_up, wq_down, wscales);
    gemm1_kernel<<<tokens / TOK1, 256, 0, stream>>>(x, wq_up, wscales, h2q, rowmax);
    gemm2_kernel<<<tokens / TOK2, 256, 0, stream>>>(h2q, wq_down, wscales, rowmax, out);
}
